// UnifiedCTNN_37873021616274
// MI455X (gfx1250) — matrix operand order repaired and checked by reading, not compiled, not run on hardware
//
#include <hip/hip_runtime.h>
#include <hip/hip_bf16.h>

typedef __attribute__((ext_vector_type(16))) _Float16 v16h;
typedef __attribute__((ext_vector_type(8)))  _Float16 v8h;
typedef __attribute__((ext_vector_type(2)))  _Float16 v2h;
typedef __attribute__((ext_vector_type(8)))  float    v8f;

#define NB 512
#define NN 64
#define ND 2
#define NH 32
#define EH 32
#define NSTEPS 2
#define JHH 16
#define T 512                    // threads per block (16 wave32 waves)
#define NWAVE (T / 32)
#define NTILES 256               // 4096 edges / 16 rows per tile
#define TPW (NTILES / NWAVE)     // 16 tiles per wave

struct alignas(16) SMem {
  _Float16 stag[NWAVE][16][32];  // per-wave staging, K-interleaved pairs
  float hv[NN * NH];             // node features (8 KB)
  float sv[NN * EH];             // v = hv @ rho_ve.T ; reused for mv2 (8 KB)
  float sc[NN * EH];             // a / (c + b1) / h1 scratch (8 KB)
  float sd[NN * EH];             // b / d scratch (8 KB)
  float mv[NN * NH];             // per-node edge-sum accumulator (8 KB)
  _Float16 wB1[32 * 32];         // eu_W1[:, :32] f16, K-permuted B operand
  _Float16 wB2[32 * 32];         // eu_W2 f16, K-permuted B operand
  float sx[NN][ND];
  float eW0[EH];
  float fin[NH + EH + 1];
  float h1[JHH];
  float hesum[EH];
  float red[4];                  // [0]=rdist sum, [1]=x^2 sum, [2]=cusp sum
  int   ss[NN];
};

// K-permutation: staging word q of a row holds logical features (q, q+16).
// Hardware K-slot s (0..31) <-> logical k, identical on A and B operands:
//   k(s) = 8*(s>>4) + ((s&15)>>1) + 16*(s&1)
__device__ inline int kperm(int s) {
  return ((s >> 4) << 3) + ((s & 15) >> 1) + ((s & 1) << 4);
}

// Branch-free tanh: native V_TANH_F32 on CDNA5.
__device__ inline float fast_tanh(float x) {
#if __has_builtin(__builtin_amdgcn_tanhf)
  return __builtin_amdgcn_tanhf(x);
#else
  const float e = __builtin_amdgcn_exp2f(x * 2.8853900817779268f);
  return 1.0f - 2.0f / (e + 1.0f);
#endif
}

// Pack two f32 into packed 2xf16 (single v_cvt_pk_rtz_f16_f32 when available).
__device__ inline unsigned int pack_h2(float a, float b) {
#if __has_builtin(__builtin_amdgcn_cvt_pkrtz)
  typedef __attribute__((ext_vector_type(2))) __fp16 v2fp16;
  union { v2fp16 h; unsigned int u; } cv;
  cv.h = __builtin_amdgcn_cvt_pkrtz(a, b);
  return cv.u;
#else
  union { v2h h; unsigned int u; } cv;
  cv.h[0] = (_Float16)a;
  cv.h[1] = (_Float16)b;
  return cv.u;
#endif
}

// A 16x32 f16 (ISA 7.12.2, wave32): lanes 0-15 row M=lane, hw slots
// {0..7,16..23}; lanes 16-31 same M, slots {8..15,24..31}. With the
// interleaved staging layout each lane's 16 values are one contiguous 32B run.
__device__ inline v16h ldsA(const _Float16* rows, int lane) {
  const int m  = lane & 15;
  const int eb = (lane & 16) ? 16 : 0;   // element base within row
  v8h lo = *(const v8h*)(rows + m * 32 + eb);
  v8h hi = *(const v8h*)(rows + m * 32 + eb + 8);
  v16h r;
#pragma unroll
  for (int q = 0; q < 8; ++q) { r[q] = lo[q]; r[q + 8] = hi[q]; }
  return r;
}

// B 32x16 f16, element order as in A: lanes 0-15 col N=lane load wB slots 0..15, lanes 16-31 wB slots
// 16..31. wB is stored pre-permuted, so loads are contiguous.
__device__ inline v16h ldsB(const _Float16* W, int nhalf, int lane) {
  const int n  = nhalf * 16 + (lane & 15);
  const int kb = (lane & 16) ? 16 : 0;
  v8h lo = *(const v8h*)(W + n * 32 + kb);
  v8h hi = *(const v8h*)(W + n * 32 + kb + 8);
  v16h r;
#pragma unroll
  for (int q = 0; q < 8; ++q) { r[q] = lo[q]; r[q + 8] = hi[q]; }
  return r;
}

__device__ inline v8f wmma_f16(v16h a, v16h b, v8f c) {
  return __builtin_amdgcn_wmma_f32_16x16x32_f16(false, a, false, b, (short)0, c,
                                                false, false);
}

__global__ __launch_bounds__(T) void ctnn_fused_kernel(
    const float* __restrict__ x, const int* __restrict__ spin,
    const float* __restrict__ node_W, const float* __restrict__ edge_W,
    const float* __restrict__ rho_ve_W, const float* __restrict__ eu_W1,
    const float* __restrict__ eu_b1, const float* __restrict__ eu_W2,
    const float* __restrict__ eu_b2, const float* __restrict__ rho_ev_W,
    const float* __restrict__ nu_W1, const float* __restrict__ nu_b1,
    const float* __restrict__ nu_W2, const float* __restrict__ nu_b2,
    const float* __restrict__ dx_W, const float* __restrict__ bf_scale_raw,
    const float* __restrict__ f_W1, const float* __restrict__ f_b1,
    const float* __restrict__ f_W2, const float* __restrict__ f_b2,
    float* __restrict__ out) {
  __shared__ SMem s;
  const int t = threadIdx.x;
  const int b = blockIdx.x;
  const int lane = t & 31;
  const int w = t >> 5;

  // ---------- Phase 0: stage x, spin, edge_W r-column ----------
  for (int u = t; u < NN * ND; u += T) s.sx[u >> 1][u & 1] = x[b * NN * ND + u];
  for (int u = t; u < NN; u += T) s.ss[u] = spin[u];
  if (t < EH) s.eW0[t] = edge_W[t * 65];
  __syncthreads();

  // ---------- Phase 1: h_v = x @ node_W.T (OMEGA=1 so x_sc = x) ----------
  for (int u = t; u < NN * NH; u += T) {
    const int i = u >> 5, k = u & 31;
    s.hv[u] = s.sx[i][0] * node_W[k * 2] + s.sx[i][1] * node_W[k * 2 + 1];
  }
  __syncthreads();

  // ---------- Phase 2: a[i] = hv@Wi.T (-> sc), b[j] = hv@Wj.T (-> sd) ----------
  for (int u = t; u < 2 * NN * EH; u += T) {
    const int half = (u >= NN * EH);
    const int v0 = u - half * NN * EH;
    const int i = v0 >> 5, k = v0 & 31;
    const float* W = &edge_W[k * 65 + 1 + half * NH];
    const float* h = &s.hv[i * NH];
    float acc = 0.f;
#pragma unroll
    for (int m = 0; m < NH; ++m) acc += h[m] * W[m];
    if (half) s.sd[v0] = acc; else s.sc[v0] = acc;
  }
  __syncthreads();

  // ---------- Phase 3: edge init straight into A-layout f16 registers ----------
  v16h hreg[TPW];
#pragma unroll
  for (int tt = 0; tt < TPW; ++tt) {
    const int gt = w * TPW + tt;       // global tile: 16 edges, fixed i
    const int i = gt >> 2;
    const int jb = (gt & 3) << 4;
    const int m = lane & 15;
    const int j = jb + m;
    const float d0 = s.sx[i][0] - s.sx[j][0];
    const float d1 = s.sx[i][1] - s.sx[j][1];
    const float r1 = sqrtf(d0 * d0 + d1 * d1 + 1e-12f);
    const int kb8 = (lane & 16) ? 8 : 0;
    v16h h;
#pragma unroll
    for (int c = 0; c < 16; ++c) {
      // component c of this lane holds logical feature k (permuted K order)
      const int k = kb8 + (c >> 1) + ((c & 1) << 4);
      h[c] = (_Float16)(r1 * s.eW0[k] + s.sc[i * 32 + k] + s.sd[j * 32 + k]);
    }
    hreg[tt] = h;
  }

  // ---------- Message-passing steps ----------
  for (int st = 0; st < NSTEPS; ++st) {
    __syncthreads();
    // v = hv @ rho_ve.T ; zero mv/hesum ; weights -> permuted f16 B operands
    for (int u = t; u < NN * EH; u += T) {
      const int i = u >> 5, k = u & 31;
      const float* h = &s.hv[i * NH];
      const float* W = &rho_ve_W[st * EH * NH + k * NH];
      float acc = 0.f;
#pragma unroll
      for (int m = 0; m < NH; ++m) acc += h[m] * W[m];
      s.sv[u] = acc;
      s.mv[u] = 0.f;
    }
    for (int u = t; u < 32 * 32; u += T) {
      const int n = u >> 5, slot = u & 31;
      const int k = kperm(slot);
      s.wB1[u] = (_Float16)eu_W1[(st * EH + n) * 96 + k];
      s.wB2[u] = (_Float16)eu_W2[st * EH * EH + n * EH + k];
    }
    if (t < EH) s.hesum[t] = 0.f;
    __syncthreads();
    // c[i][n] = b1[n] + v[i]@W1_vi.T ; d[j][n] = v[j]@W1_vj.T
    for (int u = t; u < 2 * NN * EH; u += T) {
      const int half = (u >= NN * EH);
      const int v0 = u - half * NN * EH;
      const int i = v0 >> 5, n = v0 & 31;
      const float* W = &eu_W1[(st * EH + n) * 96 + 32 + half * 32];
      const float* vp = &s.sv[i * EH];
      float acc = half ? 0.f : eu_b1[st * EH + n];
#pragma unroll
      for (int k = 0; k < EH; ++k) acc += vp[k] * W[k];
      if (half) s.sd[v0] = acc; else s.sc[v0] = acc;
    }
    __syncthreads();

    const v16h B10 = ldsB(s.wB1, 0, lane);
    const v16h B11 = ldsB(s.wB1, 1, lane);
    const v16h B20 = ldsB(s.wB2, 0, lane);
    const v16h B21 = ldsB(s.wB2, 1, lane);
    const float b20v = eu_b2[st * EH + (lane & 15)];
    const float b21v = eu_b2[st * EH + 16 + (lane & 15)];
    const int nl = lane & 15;
    const int mh = lane >> 4;
    _Float16* rows = &s.stag[w][0][0];
    unsigned int* wrows = (unsigned int*)rows;  // 16 words per row

#pragma unroll
    for (int tt = 0; tt < TPW; ++tt) {
      const int gt = w * TPW + tt;
      const int i = gt >> 2;
      const int jb = (gt & 3) << 4;
      v8f z = {};
      // GEMM1: h_e @ W1_e.T  (K=32, one WMMA per 16-col half)
      v8f a0 = wmma_f16(hreg[tt], B10, z);
      v8f a1 = wmma_f16(hreg[tt], B11, z);
      // epilogue: + c[i] + d[j] + b1, tanh; pack (n, n+16) -> one b32 store
#pragma unroll
      for (int r = 0; r < 8; ++r) {
        const int m = r + (mh << 3);   // ISA C-layout: vgpr r, lane half -> M
        const int j = jb + m;
        const float t0 = fast_tanh(a0[r] + s.sc[i * 32 + nl] + s.sd[j * 32 + nl]);
        const float t1 = fast_tanh(a1[r] + s.sc[i * 32 + 16 + nl] +
                                   s.sd[j * 32 + 16 + nl]);
        wrows[m * 16 + nl] = pack_h2(t0, t1);
      }
      asm volatile("s_wait_dscnt 0" ::: "memory");
      const v16h A2 = ldsA(rows, lane);
      // GEMM2: tanh(...) @ W2.T
      a0 = wmma_f16(A2, B20, z);
      a1 = wmma_f16(A2, B21, z);
#pragma unroll
      for (int r = 0; r < 8; ++r) {
        const int m = r + (mh << 3);
        wrows[m * 16 + nl] = pack_h2(a0[r] + b20v, a1[r] + b21v);
      }
      asm volatile("s_wait_dscnt 0" ::: "memory");
      hreg[tt] = ldsA(rows, lane);     // new h_e back to A-layout registers
      // column sums: lane == logical feature k; sum the 16 j's of this tile
      float colsum = 0.f, trisum = 0.f;
      const int ecol = 2 * (lane & 15) + (lane >> 4);  // interleaved position
#pragma unroll
      for (int m = 0; m < 16; ++m) {
        const float vv = (float)rows[m * 32 + ecol];
        colsum += vv;
        if (jb + m > i) trisum += vv;  // upper-triangular (i<j) edges
      }
      atomicAdd(&s.mv[i * 32 + lane], colsum);
      if (st == NSTEPS - 1) atomicAdd(&s.hesum[lane], trisum);
    }
    __syncthreads();

    // node update: mv2 = mv @ rho_ev.T -> sv
    for (int u = t; u < NN * NH; u += T) {
      const int i = u >> 5, k = u & 31;
      const float* W = &rho_ev_W[st * NH * EH + k * EH];
      const float* mp = &s.mv[i * EH];
      float acc = 0.f;
#pragma unroll
      for (int m = 0; m < EH; ++m) acc += mp[m] * W[m];
      s.sv[u] = acc;
    }
    __syncthreads();
    // h1 = tanh(nu_W1 @ [hv, mv2] + b1) -> sc
    for (int u = t; u < NN * NH; u += T) {
      const int i = u >> 5, k = u & 31;
      const float* W = &nu_W1[(st * NH + k) * 64];
      const float* h = &s.hv[i * NH];
      const float* m2 = &s.sv[i * NH];
      float acc = nu_b1[st * NH + k];
#pragma unroll
      for (int m = 0; m < NH; ++m) acc += h[m] * W[m] + m2[m] * W[32 + m];
      s.sc[u] = fast_tanh(acc);
    }
    __syncthreads();
    // hv = nu_W2 @ h1 + b2
    for (int u = t; u < NN * NH; u += T) {
      const int i = u >> 5, k = u & 31;
      const float* W = &nu_W2[st * NH * NH + k * NH];
      const float* h = &s.sc[i * NH];
      float acc = nu_b2[st * NH + k];
#pragma unroll
      for (int m = 0; m < NH; ++m) acc += h[m] * W[m];
      s.hv[u] = acc;
    }
  }
  __syncthreads();

  // ---------- Readout ----------
  if (t < 4) s.red[t] = 0.f;
  if (t < NH) {
    float a = 0.f;
    for (int i = 0; i < NN; ++i) a += s.hv[i * NH + t];
    s.fin[t] = a * (1.f / 64.f);
  }
  if (t >= 32 && t < 64) s.fin[t] = s.hesum[t - 32] * (1.f / 2016.f);
  // dx = softplus(raw) * tanh(hv @ dx_W.T)
  for (int u = t; u < NN * ND; u += T) {
    const int i = u >> 1, dd = u & 1;
    float acc = 0.f;
#pragma unroll
    for (int k = 0; k < NH; ++k) acc += s.hv[i * NH + k] * dx_W[dd * NH + k];
    const float scale = log1pf(expf(bf_scale_raw[0]));
    out[b * NN * ND + u] = scale * fast_tanh(acc);
  }
  __syncthreads();

  float rsum = 0.f, csum = 0.f;
  for (int e = t; e < NN * NN; e += T) {
    const int i = e >> 6, j = e & 63;
    if (i != j) {
      const float d0 = s.sx[i][0] - s.sx[j][0];
      const float d1 = s.sx[i][1] - s.sx[j][1];
      const float r2v = d0 * d0 + d1 * d1;
      rsum += sqrtf(r2v + 1e-12f);
      if (i < j) {
        const float rp = sqrtf(r2v + 1e-30f);
        const float g = (s.ss[i] == s.ss[j]) ? (1.f / 3.f) : 1.f;  // D=2
        csum += g * rp / (1.f + rp);                               // cusp_len=1
      }
    }
  }
  atomicAdd(&s.red[0], rsum);
  atomicAdd(&s.red[2], csum);
  if (t < NN)
    atomicAdd(&s.red[1], s.sx[t][0] * s.sx[t][0] + s.sx[t][1] * s.sx[t][1]);
  __syncthreads();
  if (t == 0) s.fin[64] = s.red[0] * (1.f / (64.f * 63.f));
  __syncthreads();
  if (t < JHH) {
    float a = f_b1[t];
    for (int m = 0; m < NH + EH + 1; ++m) a += s.fin[m] * f_W1[t * 65 + m];
    s.h1[t] = fast_tanh(a);
  }
  __syncthreads();
  if (t == 0) {
    float fnn = f_b2[0];
#pragma unroll
    for (int m = 0; m < JHH; ++m) fnn += s.h1[m] * f_W2[m];
    const float env = expf(-0.5f * s.red[1]);   // ell2 = 1
    out[NB * NN * ND + b] = fnn * env + 0.2f * s.red[2];
  }
}

extern "C" void kernel_launch(void* const* d_in, const int* in_sizes, int n_in,
                              void* d_out, int out_size, void* d_ws,
                              size_t ws_size, hipStream_t stream) {
  (void)in_sizes; (void)n_in; (void)out_size; (void)d_ws; (void)ws_size;
  ctnn_fused_kernel<<<NB, T, 0, stream>>>(
      (const float*)d_in[0],  (const int*)d_in[1],   (const float*)d_in[2],
      (const float*)d_in[3],  (const float*)d_in[4], (const float*)d_in[5],
      (const float*)d_in[6],  (const float*)d_in[7], (const float*)d_in[8],
      (const float*)d_in[9],  (const float*)d_in[10],(const float*)d_in[11],
      (const float*)d_in[12], (const float*)d_in[13],(const float*)d_in[14],
      (const float*)d_in[15], (const float*)d_in[16],(const float*)d_in[17],
      (const float*)d_in[18], (const float*)d_in[19],(float*)d_out);
}
